// ResidualSchedulingGNN_63840393888599
// MI455X (gfx1250) — compile-verified, matrix operand order checked by reading, not run on hardware
//
#include <hip/hip_runtime.h>

// MI455X / gfx1250, wave32. All GEMMs via v_wmma_f32_16x16x32_bf16.
// ~450 GFLOP of GEMM, ~8 GB traffic -> balanced vs 23.3 TB/s HBM.
// GEMM: 128x128x32 tiles, 8 waves (4x2), wave tile 32x64 (2x4 WMMA accums),
// double-buffered LDS with register prefetch so HBM loads overlap the WMMAs.

typedef __attribute__((ext_vector_type(16))) __bf16 v16bf;
typedef __attribute__((ext_vector_type(8)))  __bf16 v8bf;
typedef __attribute__((ext_vector_type(4)))  __bf16 v4bf;
typedef __attribute__((ext_vector_type(8)))  float  v8f;

#define TPB 256
#define LDS_STRIDE 40   // bf16 elems per LDS row: 80B, 16B-aligned, bank-spread
#define LDS_TILE (128 * LDS_STRIDE)

// ---------------------------------------------------------------- utilities

__global__ void copy_f4_kernel(const float* __restrict__ src,
                               float* __restrict__ dst, long n4) {
  long i = (long)blockIdx.x * blockDim.x + threadIdx.x;
  if (i < n4) ((float4*)dst)[i] = ((const float4*)src)[i];
}

// out[dst[e], :] += srcX[src[e], :]   (H=256, float4 per thread-chunk)
__global__ void scatter_add_kernel(const float* __restrict__ srcX,
                                   const int* __restrict__ src,
                                   const int* __restrict__ dst,
                                   float* __restrict__ out, int E) {
  long gid = (long)blockIdx.x * blockDim.x + threadIdx.x;
  int e = (int)(gid >> 6);
  if (e >= E) return;
  int c = (int)(gid & 63) << 2;
  const float4 v = *(const float4*)(srcX + (size_t)src[e] * 256 + c);
  float* o = out + (size_t)dst[e] * 256 + c;
  atomicAdd(o + 0, v.x);
  atomicAdd(o + 1, v.y);
  atomicAdd(o + 2, v.z);
  atomicAdd(o + 3, v.w);
}

// WT[n*K+k] = (bf16) W[k*N+n]  (weights are tiny; coalesced writes)
__global__ void wtrans_kernel(const float* __restrict__ W,
                              __bf16* __restrict__ WT, int K, int N) {
  int gid = blockIdx.x * blockDim.x + threadIdx.x;
  if (gid >= K * N) return;
  int n = gid / K, k = gid - n * K;
  WT[gid] = (__bf16)W[(size_t)k * N + n];
}

// Deterministic column statistics: stage 1 -> per-chunk partial sum/sumsq
__global__ void colstats_part_kernel(const float* __restrict__ X, int M, int C,
                                     float* __restrict__ partS,
                                     float* __restrict__ partQ) {
  int col = threadIdx.x;  // blockDim.x == C
  float s = 0.f, q = 0.f;
  for (int r = blockIdx.x; r < M; r += gridDim.x) {
    float v = X[(size_t)r * C + col];
    s += v; q += v * v;
  }
  partS[(size_t)blockIdx.x * 512 + col] = s;
  partQ[(size_t)blockIdx.x * 512 + col] = q;
}

// stage 2: fixed-order reduction + fold BN params into (scale, shift)
__global__ void colstats_fin_kernel(const float* __restrict__ partS,
                                    const float* __restrict__ partQ,
                                    const float* __restrict__ g,
                                    const float* __restrict__ be,
                                    float invM, int C, int nchunk,
                                    float* __restrict__ scale,
                                    float* __restrict__ shift) {
  int col = threadIdx.x;
  if (col >= C) return;
  float s = 0.f, q = 0.f;
  for (int b = 0; b < nchunk; ++b) {
    s += partS[(size_t)b * 512 + col];
    q += partQ[(size_t)b * 512 + col];
  }
  float mean = s * invM;
  float var  = q * invM - mean * mean;
  float sc = g[col] * rsqrtf(var + 1e-5f);
  scale[col] = sc;
  shift[col] = be[col] - mean * sc;
}

// out[p] = dot(relu(z2[p]*scale+shift), w3) + b3   (one wave32 per row)
__global__ void score_final_kernel(const float* __restrict__ Z,
                                   const float* __restrict__ scale,
                                   const float* __restrict__ shift,
                                   const float* __restrict__ w3,
                                   const float* __restrict__ b3,
                                   float* __restrict__ out, int P) {
  int wv   = threadIdx.x >> 5;
  int lane = threadIdx.x & 31;
  int p = blockIdx.x * 8 + wv;
  if (p >= P) return;
  int k = lane * 4;
  float4 z  = *(const float4*)(Z + (size_t)p * 128 + k);
  float4 sc = *(const float4*)(scale + k);
  float4 sh = *(const float4*)(shift + k);
  float4 w  = *(const float4*)(w3 + k);
  float s = 0.f;
  s += fmaxf(z.x * sc.x + sh.x, 0.f) * w.x;
  s += fmaxf(z.y * sc.y + sh.y, 0.f) * w.y;
  s += fmaxf(z.z * sc.z + sh.z, 0.f) * w.z;
  s += fmaxf(z.w * sc.w + sh.w, 0.f) * w.w;
#pragma unroll
  for (int m = 16; m >= 1; m >>= 1) s += __shfl_xor(s, m, 32);
  if (lane == 0) out[p] = s + b3[0];
}

// ---------------------------------------------------------------- WMMA GEMM
// C[M,N] = prologue(A[M,K]) * W[K,N] + bias (+C) (+addbuf)
// AMODE: 0 = plain fp32 A, 1 = relu(A*scale[k]+shift[k]) (fused BN+ReLU),
//        2 = pair-gather concat [xm[pm[i]] | xop[po[i]]] (K = 512)
// BT = bf16 weights pre-transposed to [N][K].
template <int AMODE>
__global__ __launch_bounds__(TPB) void gemm_bf16_kernel(
    const float* __restrict__ A, int M, int K,
    const __bf16* __restrict__ BT, const float* __restrict__ bias,
    const float* __restrict__ scale, const float* __restrict__ shift,
    const float* __restrict__ xm, const float* __restrict__ xop,
    const int* __restrict__ pm, const int* __restrict__ po,
    float* __restrict__ C, int N, const float* __restrict__ addbuf,
    int accumulate) {
  __shared__ __bf16 As[2 * LDS_TILE];  // double buffered
  __shared__ __bf16 Bs[2 * LDS_TILE];

  const int t = threadIdx.x;
  const int blockM = blockIdx.x * 128;
  const int blockN = blockIdx.y * 128;

  const int wv = t >> 5;
  const int lane = t & 31;
  const int hl = lane >> 4;   // K-half selector per ISA 16-bit layout
  const int r  = lane & 15;   // row (A) / col (B) within 16
  const int wm = (wv >> 1) * 32;
  const int wn = (wv & 1) * 64;

  // staging coords
  const int ar  = t >> 3;       // A: 32 rows per pass, 4 passes
  const int ac4 = (t & 7) * 4;  // A: 4 cols per thread
  const int bn  = t >> 1;       // B: one n-row per 2 threads
  const int bk  = (t & 1) * 16; // B: 16 k per thread

  v8f acc[2][4];
#pragma unroll
  for (int i = 0; i < 2; ++i)
#pragma unroll
    for (int j = 0; j < 4; ++j)
#pragma unroll
      for (int e = 0; e < 8; ++e) acc[i][j][e] = 0.0f;

  // Register prefetch state (filled by gload, drained to LDS by sstore).
  float4 apf[4];
  v8bf bpf0, bpf1;
  float4 pf_sc, pf_sh;

  // Batched, branch-free global loads for k-tile kt (row index clamped so
  // all 4 A loads + 2 B loads issue back-to-back; OOB rows zeroed at store).
  auto gload = [&](int kt) {
    const int kk = kt * 32;
    const int kg = kk + ac4;
    const float* asrc;
    int kadj = kg;
    const int* pidx = pm;
    if (AMODE == 2) {
      const bool lowhalf = (kk < 256);   // uniform per k-tile (256 % 32 == 0)
      asrc = lowhalf ? xm : xop;
      pidx = lowhalf ? pm : po;
      kadj = lowhalf ? kg : kg - 256;
    } else {
      asrc = A;
    }
#pragma unroll
    for (int i = 0; i < 4; ++i) {
      const int gm = blockM + ar + i * 32;
      const int gmc = gm < M ? gm : M - 1;  // clamp: always a valid load
      if (AMODE == 2)
        apf[i] = *(const float4*)(asrc + (size_t)pidx[gmc] * 256 + kadj);
      else
        apf[i] = *(const float4*)(asrc + (size_t)gmc * K + kadj);
    }
    const __bf16* gsrc = BT + (size_t)(blockN + bn) * K + kk + bk;
    bpf0 = *(const v8bf*)(gsrc);
    bpf1 = *(const v8bf*)(gsrc + 8);
    if (AMODE == 1) {
      pf_sc = *(const float4*)(scale + kg);
      pf_sh = *(const float4*)(shift + kg);
    }
  };

  // Convert + store prefetched tile into LDS buffer (kt & 1).
  auto sstore = [&](int kt) {
    __bf16* Asb = As + (kt & 1) * LDS_TILE;
    __bf16* Bsb = Bs + (kt & 1) * LDS_TILE;
#pragma unroll
    for (int i = 0; i < 4; ++i) {
      const int row = ar + i * 32;
      const bool ok = (blockM + row) < M;
      float4 v = apf[i];
      if (AMODE == 1) {
        v.x = fmaxf(v.x * pf_sc.x + pf_sh.x, 0.f);
        v.y = fmaxf(v.y * pf_sc.y + pf_sh.y, 0.f);
        v.z = fmaxf(v.z * pf_sc.z + pf_sh.z, 0.f);
        v.w = fmaxf(v.w * pf_sc.w + pf_sh.w, 0.f);
      }
      v.x = ok ? v.x : 0.f;   // branchless zero of OOB rows
      v.y = ok ? v.y : 0.f;
      v.z = ok ? v.z : 0.f;
      v.w = ok ? v.w : 0.f;
      v4bf h;
      h[0] = (__bf16)v.x; h[1] = (__bf16)v.y;
      h[2] = (__bf16)v.z; h[3] = (__bf16)v.w;
      *(v4bf*)&Asb[row * LDS_STRIDE + ac4] = h;
    }
    *(v8bf*)&Bsb[bn * LDS_STRIDE + bk]     = bpf0;
    *(v8bf*)&Bsb[bn * LDS_STRIDE + bk + 8] = bpf1;
  };

  const int NKT = K / 32;
  gload(0);
  sstore(0);

  for (int kt = 0; kt < NKT; ++kt) {
    __syncthreads();
    const bool more = (kt + 1) < NKT;
    if (more) gload(kt + 1);   // HBM loads fly while the WMMAs below run

    const __bf16* Asb = As + (kt & 1) * LDS_TILE;
    const __bf16* Bsb = Bs + (kt & 1) * LDS_TILE;
    v16bf af[2], bfr[4];
#pragma unroll
    for (int mt = 0; mt < 2; ++mt) {
      const int row = wm + mt * 16 + r;
      v8bf lo = *(const v8bf*)&Asb[row * LDS_STRIDE + hl * 8];
      v8bf hi = *(const v8bf*)&Asb[row * LDS_STRIDE + 16 + hl * 8];
      af[mt] = __builtin_shufflevector(lo, hi, 0, 1, 2, 3, 4, 5, 6, 7, 8, 9,
                                       10, 11, 12, 13, 14, 15);
    }
#pragma unroll
    for (int nt = 0; nt < 4; ++nt) {
      const int col = wn + nt * 16 + r;
      v8bf lo = *(const v8bf*)&Bsb[col * LDS_STRIDE + hl * 8];
      v8bf hi = *(const v8bf*)&Bsb[col * LDS_STRIDE + 16 + hl * 8];
      bfr[nt] = __builtin_shufflevector(lo, hi, 0, 1, 2, 3, 4, 5, 6, 7, 8, 9,
                                        10, 11, 12, 13, 14, 15);
    }
#pragma unroll
    for (int mt = 0; mt < 2; ++mt)
#pragma unroll
      for (int nt = 0; nt < 4; ++nt)
        acc[mt][nt] = __builtin_amdgcn_wmma_f32_16x16x32_bf16(
            false, af[mt], false, bfr[nt], (short)0, acc[mt][nt], false, false);

    if (more) sstore(kt + 1);  // drain prefetch into the other LDS buffer
  }

  // ---- epilogue: + bias (+ existing C) (+ residual) ----
#pragma unroll
  for (int mt = 0; mt < 2; ++mt) {
#pragma unroll
    for (int nt = 0; nt < 4; ++nt) {
      const int col = blockN + wn + nt * 16 + r;
      const float b = bias[col];
#pragma unroll
      for (int vv = 0; vv < 8; ++vv) {
        const int row = blockM + wm + mt * 16 + hl * 8 + vv;
        if (row < M) {
          const size_t idx = (size_t)row * N + col;
          float val = acc[mt][nt][vv] + b;
          if (accumulate) val += C[idx];
          if (addbuf) val += addbuf[idx];
          C[idx] = val;
        }
      }
    }
  }
}

// ---------------------------------------------------------------- host side

extern "C" void kernel_launch(void* const* d_in, const int* in_sizes, int n_in,
                              void* d_out, int out_size, void* d_ws,
                              size_t ws_size, hipStream_t stream) {
  const int H = 256;
  const int NOP = in_sizes[0] / H;
  const int NM  = in_sizes[1] / H;
  const int EOO = in_sizes[2];
  const int EMO = in_sizes[4];
  const int EOM = in_sizes[6];
  const int P   = in_sizes[8];

  const float* x_op0 = (const float*)d_in[0];
  const float* x_m0  = (const float*)d_in[1];
  const int* eoo_s = (const int*)d_in[2];
  const int* eoo_d = (const int*)d_in[3];
  const int* emo_s = (const int*)d_in[4];
  const int* emo_d = (const int*)d_in[5];
  const int* eom_s = (const int*)d_in[6];
  const int* eom_d = (const int*)d_in[7];
  const int* pm = (const int*)d_in[8];
  const int* po = (const int*)d_in[9];

  struct MlpP { const float *w1, *b1, *g1, *be1, *w2, *b2; };
  MlpP conv[3][3];
  int idx = 10;
  for (int l = 0; l < 3; ++l)
    for (int e = 0; e < 3; ++e) {
      conv[l][e].w1  = (const float*)d_in[idx++];
      conv[l][e].b1  = (const float*)d_in[idx++];
      conv[l][e].g1  = (const float*)d_in[idx++];
      conv[l][e].be1 = (const float*)d_in[idx++];
      conv[l][e].w2  = (const float*)d_in[idx++];
      conv[l][e].b2  = (const float*)d_in[idx++];
    }
  const float* sw1  = (const float*)d_in[idx++];
  const float* sb1  = (const float*)d_in[idx++];
  const float* sg1  = (const float*)d_in[idx++];
  const float* sbe1 = (const float*)d_in[idx++];
  const float* sw2  = (const float*)d_in[idx++];
  const float* sb2  = (const float*)d_in[idx++];
  const float* sg2  = (const float*)d_in[idx++];
  const float* sbe2 = (const float*)d_in[idx++];
  const float* sw3  = (const float*)d_in[idx++];
  const float* sb3  = (const float*)d_in[idx++];

  // ---- workspace carve-up (256B aligned) ----
  char* wsb = (char*)d_ws;
  size_t off = 0;
  auto take = [&](size_t bytes) -> char* {
    char* p = wsb + off;
    off = (off + bytes + 255) & ~(size_t)255;
    return p;
  };
  size_t nopH = (size_t)NOP * H, nmH = (size_t)NM * H;
  size_t inElems = nopH > (size_t)P * (H / 2) ? nopH : (size_t)P * (H / 2);
  size_t zElems  = (size_t)(P > NOP ? P : NOP) * H;
  float* xopA  = (float*)take(nopH * 4);
  float* xopB  = (float*)take(nopH * 4);
  float* xmA   = (float*)take(nmH * 4);
  float* xmB   = (float*)take(nmH * 4);
  float* inbuf = (float*)take(inElems * 4);  // gather buf; reused as z2 [P,128]
  float* zbuf  = (float*)take(zElems * 4);   // z1 buffers
  float* partS = (float*)take((size_t)1024 * 512 * 4);
  float* partQ = (float*)take((size_t)1024 * 512 * 4);
  float* scale = (float*)take(512 * 4);
  float* shift = (float*)take(512 * 4);
  __bf16* WT   = (__bf16*)take((size_t)512 * 256 * 2);

  const int NCHUNK = 1024;

  auto run_gemm = [&](int amode, const float* Ap, int M, int K, const float* W,
                      const float* bias, const float* gxm, const float* gxop,
                      float* Cp, int N, const float* addb, int accum) {
    int tot = K * N;
    wtrans_kernel<<<(tot + 255) / 256, 256, 0, stream>>>(W, WT, K, N);
    dim3 grid((M + 127) / 128, N / 128);
    if (amode == 0)
      gemm_bf16_kernel<0><<<grid, TPB, 0, stream>>>(
          Ap, M, K, WT, bias, scale, shift, gxm, gxop, pm, po, Cp, N, addb, accum);
    else if (amode == 1)
      gemm_bf16_kernel<1><<<grid, TPB, 0, stream>>>(
          Ap, M, K, WT, bias, scale, shift, gxm, gxop, pm, po, Cp, N, addb, accum);
    else
      gemm_bf16_kernel<2><<<grid, TPB, 0, stream>>>(
          Ap, M, K, WT, bias, scale, shift, gxm, gxop, pm, po, Cp, N, addb, accum);
  };

  auto run_stats = [&](const float* X, int M, int Cc, const float* g,
                       const float* be) {
    colstats_part_kernel<<<NCHUNK, Cc, 0, stream>>>(X, M, Cc, partS, partQ);
    colstats_fin_kernel<<<1, Cc, 0, stream>>>(partS, partQ, g, be,
                                              1.0f / (float)M, Cc, NCHUNK,
                                              scale, shift);
  };

  // GIN MLP: z = in@w1+b1 ; stats(z) ; out (+)= relu(bn(z))@w2+b2 (+resid)
  auto run_mlp = [&](const float* in, int M, const MlpP& p, float* outp,
                     int accum, const float* resid) {
    run_gemm(0, in, M, H, p.w1, p.b1, nullptr, nullptr, zbuf, H, nullptr, 0);
    run_stats(zbuf, M, H, p.g1, p.be1);
    run_gemm(1, zbuf, M, H, p.w2, p.b2, nullptr, nullptr, outp, H, resid, accum);
  };

  auto copy_buf = [&](const float* s, float* d, size_t elems) {
    long n4 = (long)(elems / 4);
    copy_f4_kernel<<<(unsigned)((n4 + 255) / 256), 256, 0, stream>>>(s, d, n4);
  };
  auto scatter = [&](const float* srcX, const int* si, const int* di,
                     float* outp, int E) {
    long thr = (long)E * 64;
    scatter_add_kernel<<<(unsigned)((thr + 255) / 256), 256, 0, stream>>>(
        srcX, si, di, outp, E);
  };

  // ---- 3 hetero-GIN layers (ping-pong node features) ----
  const float* cur_op = x_op0;
  const float* cur_m  = x_m0;
  float* outs_op[3] = {xopA, xopB, xopA};
  float* outs_m[3]  = {xmA, xmB, xmA};
  for (int l = 0; l < 3; ++l) {
    float* nop = outs_op[l];
    float* nm  = outs_m[l];
    const float* res_op = (l > 0) ? cur_op : nullptr;
    const float* res_m  = (l > 0) ? cur_m : nullptr;

    // (op, precedes, op)
    copy_buf(cur_op, inbuf, nopH);
    scatter(cur_op, eoo_s, eoo_d, inbuf, EOO);
    run_mlp(inbuf, NOP, conv[l][0], nop, 0, res_op);

    // (machine, executes, op)  -> accumulate into same op output
    copy_buf(cur_op, inbuf, nopH);
    scatter(cur_m, emo_s, emo_d, inbuf, EMO);
    run_mlp(inbuf, NOP, conv[l][1], nop, 1, nullptr);

    // (op, runs_on, machine)
    copy_buf(cur_m, inbuf, nmH);
    scatter(cur_op, eom_s, eom_d, inbuf, EOM);
    run_mlp(inbuf, NM, conv[l][2], nm, 0, res_m);

    cur_op = nop;
    cur_m = nm;
  }

  // ---- scoring head ----
  // z1 = concat(xm[pm], xop[po]) @ w1 + b1  (gather fused into GEMM A-load)
  run_gemm(2, nullptr, P, 2 * H, sw1, sb1, cur_m, cur_op, zbuf, H, nullptr, 0);
  run_stats(zbuf, P, H, sg1, sbe1);
  // z2 = relu(bn(z1)) @ w2 + b2
  run_gemm(1, zbuf, P, H, sw2, sb2, nullptr, nullptr, inbuf, H / 2, nullptr, 0);
  run_stats(inbuf, P, H / 2, sg2, sbe2);
  // out = relu(bn(z2)) @ w3 + b3
  score_final_kernel<<<(P + 7) / 8, 256, 0, stream>>>(
      inbuf, scale, shift, sw3, sb3, (float*)d_out, P);
}
